// AttnBlock_11321533792900
// MI455X (gfx1250) — matrix operand order repaired and checked by reading, not compiled, not run on hardware
//
#include <hip/hip_runtime.h>
#include <hip/hip_bf16.h>
#include <stdint.h>

// ---------------------------------------------------------------------------
// AttnBlock for MI455X (gfx1250): LN -> QKV GEMM (bf16 WMMA) -> flash attn
// (bf16 WMMA, fp32 accum, online softmax) -> output GEMM (fp32 out).
// Tile movement via Tensor Data Mover (TENSOR_LOAD_TO_LDS) when available.
// B=2, L=2048, EMB=1024, HEADS=16, HEAD_SIZE=64.
// ---------------------------------------------------------------------------

typedef __bf16 bf16_t;
typedef __attribute__((ext_vector_type(16))) __bf16 v16bf;
typedef __attribute__((ext_vector_type(8)))  float  v8f;

#if defined(__AMDGCN__) && __has_builtin(__builtin_amdgcn_tensor_load_to_lds) && \
    __has_builtin(__builtin_amdgcn_s_wait_tensorcnt)
#define ATTN_USE_TDM 1
#else
#define ATTN_USE_TDM 0
#endif

#if ATTN_USE_TDM
typedef __attribute__((ext_vector_type(4))) unsigned int u32x4;
typedef __attribute__((ext_vector_type(8))) int          i32x8;
typedef __attribute__((ext_vector_type(4))) int          i32x4;

__device__ __forceinline__ unsigned lds_offset_of(const void* p) {
  // Generic shared address: aperture in [63:32], DS byte offset in [31:0].
  return (unsigned)(unsigned long long)(size_t)p;
}

// 2D tile DMA global->LDS via Tensor Data Mover (D# per CDNA5 ISA 8.3-8.6).
// data_size=1 (2-byte elements). LDS padding done by the DMA engine:
//   pad_interval: (bytes_per_row >> 3) == (1 << pad_interval)
//   pad_amount:   pads (pad_amount+1) DWORDs after each interval.
__device__ __forceinline__ void tdm_load_tile_2d(
    const bf16_t* gsrc, unsigned lds_off, unsigned row_stride_elems,
    unsigned tile_w, unsigned tile_h, unsigned pad_interval,
    unsigned pad_amount) {
  unsigned long long ga = (unsigned long long)(size_t)gsrc;
  u32x4 g0;
  g0[0] = 1u;                                     // count=1 (valid user D#)
  g0[1] = lds_off;                                // lds_addr (bytes)
  g0[2] = (unsigned)ga;                           // global_addr[31:0]
  g0[3] = (unsigned)((ga >> 32) & 0x01ffffffu) |  // global_addr[56:32]
          (2u << 30);                             // type=2 ("image")
  i32x8 g1;
  g1[0] = (int)((1u << 16) |                      // data_size=1 (2B)
                (1u << 20) |                      // pad_enable
                (pad_interval << 22) | (pad_amount << 25));
  g1[1] = (int)((row_stride_elems & 0xffffu) << 16);          // tensor_dim0 lo
  g1[2] = (int)(((row_stride_elems >> 16) & 0xffffu) |
                ((tile_h & 0xffffu) << 16));                  // dim0 hi|dim1 lo
  g1[3] = (int)(((tile_h >> 16) & 0xffffu) | (tile_w << 16)); // dim1 hi|tile_dim0
  g1[4] = (int)(tile_h & 0xffffu);                            // tile_dim1
  g1[5] = (int)row_stride_elems;                              // dim0_stride lo32
  g1[6] = 0;
  g1[7] = 0;
  i32x4 zz4 = {0, 0, 0, 0};
  i32x8 zz8 = {0, 0, 0, 0, 0, 0, 0, 0};
  // 6-arg form (clang-23 / therock-10.0 headers): groups 0..4 + cpol.
  __builtin_amdgcn_tensor_load_to_lds(g0, g1, zz4, zz4, zz8, 0);
}
#endif  // ATTN_USE_TDM

__device__ __forceinline__ v8f wmma_bf16(v16bf a, v16bf b, v8f c) {
  return __builtin_amdgcn_wmma_f32_16x16x32_bf16(false, a, false, b,
                                                 (short)0, c, false, false);
}

// A fragment (16x32 bf16), row-major source, row stride ld (elements).
__device__ __forceinline__ v16bf load_frag_a(const bf16_t* p, int ld, int lane) {
  int m  = lane & 15;
  int k0 = (lane >> 4) << 4;
  const bf16_t* r = p + (size_t)m * ld;
  v16bf a;
#pragma unroll
  for (int j = 0; j < 8; ++j) {
    int kb = k0 + (j << 1);
    a[2 * j]     = r[kb];
    a[2 * j + 1] = r[kb + 1];
  }
  return a;
}

// B fragment (32x16) where B(k,n) = p[n*ld + k]  (tile row-major [n][k]).
__device__ __forceinline__ v16bf load_frag_bt(const bf16_t* p, int ld, int lane) {
  int n  = lane & 15;
  int k0 = (lane >> 4) << 4;
  const bf16_t* r = p + (size_t)n * ld + k0;
  v16bf b;
#pragma unroll
  for (int e = 0; e < 16; ++e) b[e] = r[e];
  return b;
}

// B fragment (32x16) where B(k,n) = p[k*ld + n] (row-major over k).
__device__ __forceinline__ v16bf load_frag_b(const bf16_t* p, int ld, int lane) {
  int n  = lane & 15;
  int k0 = (lane >> 4) << 4;
  v16bf b;
#pragma unroll
  for (int e = 0; e < 16; ++e) b[e] = p[(size_t)(k0 + e) * ld + n];
  return b;
}

// ---------------------------------------------------------------------------
// Kernel 1: LayerNorm (eps=1e-5) + cast to bf16. One block per token row.
// ---------------------------------------------------------------------------
__global__ __launch_bounds__(256) void ln_bf16_kernel(
    const float* __restrict__ x, const float* __restrict__ g,
    const float* __restrict__ bta, bf16_t* __restrict__ h) {
  __shared__ float red[2][8];
  int row = blockIdx.x;
  const float* xr = x + (size_t)row * 1024;
  float v[4], s = 0.f, s2 = 0.f;
#pragma unroll
  for (int i = 0; i < 4; ++i) {
    v[i] = xr[threadIdx.x + 256 * i];
    s += v[i];
    s2 += v[i] * v[i];
  }
#pragma unroll
  for (int off = 16; off > 0; off >>= 1) {
    s  += __shfl_xor(s, off, 32);
    s2 += __shfl_xor(s2, off, 32);
  }
  int wid = threadIdx.x >> 5;
  if ((threadIdx.x & 31) == 0) { red[0][wid] = s; red[1][wid] = s2; }
  __syncthreads();
  if (wid == 0) {
    float a  = (threadIdx.x < 8) ? red[0][threadIdx.x] : 0.f;
    float b2 = (threadIdx.x < 8) ? red[1][threadIdx.x] : 0.f;
#pragma unroll
    for (int off = 4; off > 0; off >>= 1) {
      a  += __shfl_xor(a, off, 32);
      b2 += __shfl_xor(b2, off, 32);
    }
    if (threadIdx.x == 0) { red[0][0] = a; red[1][0] = b2; }
  }
  __syncthreads();
  float mu  = red[0][0] * (1.f / 1024.f);
  float var = red[1][0] * (1.f / 1024.f) - mu * mu;
  float rs  = rsqrtf(var + 1e-5f);
#pragma unroll
  for (int i = 0; i < 4; ++i) {
    int c = threadIdx.x + 256 * i;
    h[(size_t)row * 1024 + c] = (bf16_t)(((v[i] - mu) * rs) * g[c] + bta[c]);
  }
}

// ---------------------------------------------------------------------------
// Kernel 2: fp32 -> bf16 elementwise (weights).
// ---------------------------------------------------------------------------
__global__ __launch_bounds__(256) void f32_to_bf16_kernel(
    const float* __restrict__ src, bf16_t* __restrict__ dst, int n) {
  int i = (blockIdx.x * 256 + threadIdx.x) * 4;
  if (i + 3 < n) {
    float4 f = *(const float4*)(src + i);
    dst[i]     = (bf16_t)f.x;
    dst[i + 1] = (bf16_t)f.y;
    dst[i + 2] = (bf16_t)f.z;
    dst[i + 3] = (bf16_t)f.w;
  }
}

// ---------------------------------------------------------------------------
// Kernel 3: Y[m,n] = sum_k A[m,k]*W[n,k] + bias[n].  128x128x32 tiles,
// 8 waves (2x4), each wave 64x32 -> 8 WMMAs/K-step. TDM double-buffered.
// ---------------------------------------------------------------------------
template <typename OutT>
__global__ __launch_bounds__(256) void gemm_bf16_kernel(
    const bf16_t* __restrict__ A, const bf16_t* __restrict__ W,
    const float* __restrict__ bias, OutT* __restrict__ Y) {
  constexpr int K = 1024, NCOL = 1024, LDA = 40;  // 80B padded LDS stride
  int m0 = blockIdx.x * 128, n0 = blockIdx.y * 128;
  int tid = threadIdx.x, lane = tid & 31, wave = tid >> 5;
  int wm = wave >> 2, wn = wave & 3;

  v8f acc[4][2];
#pragma unroll
  for (int mi = 0; mi < 4; ++mi)
#pragma unroll
    for (int ni = 0; ni < 2; ++ni) acc[mi][ni] = (v8f){};

#if ATTN_USE_TDM
  __shared__ __align__(16) bf16_t sA[2][128 * LDA];
  __shared__ __align__(16) bf16_t sB[2][128 * LDA];
  // Tile rows: 32 elems = 64B -> pad_interval=3; pad 4 DWORDs (16B = 8 elems)
  // -> pad_amount=3; DMA-written stride = 40 elems = LDA.
  if (wave == 0)
    tdm_load_tile_2d(A + (size_t)m0 * K, lds_offset_of(&sA[0][0]), K, 32, 128, 3, 3);
  else if (wave == 1)
    tdm_load_tile_2d(W + (size_t)n0 * K, lds_offset_of(&sB[0][0]), K, 32, 128, 3, 3);
  if (wave < 2) __builtin_amdgcn_s_wait_tensorcnt(0);
  __syncthreads();

  for (int it = 0; it < K / 32; ++it) {
    int buf = it & 1;
    if (it + 1 < K / 32) {  // DMA next K-tile while computing this one
      if (wave == 0)
        tdm_load_tile_2d(A + (size_t)m0 * K + (it + 1) * 32,
                         lds_offset_of(&sA[buf ^ 1][0]), K, 32, 128, 3, 3);
      else if (wave == 1)
        tdm_load_tile_2d(W + (size_t)n0 * K + (it + 1) * 32,
                         lds_offset_of(&sB[buf ^ 1][0]), K, 32, 128, 3, 3);
    }
    const bf16_t* tA = &sA[buf][0];
    const bf16_t* tB = &sB[buf][0];
    v16bf bfrag[2];
#pragma unroll
    for (int ni = 0; ni < 2; ++ni)
      bfrag[ni] = load_frag_bt(tB + (wn * 32 + ni * 16) * LDA, LDA, lane);
#pragma unroll
    for (int mi = 0; mi < 4; ++mi) {
      v16bf afrag = load_frag_a(tA + (wm * 64 + mi * 16) * LDA, LDA, lane);
#pragma unroll
      for (int ni = 0; ni < 2; ++ni)
        acc[mi][ni] = wmma_bf16(afrag, bfrag[ni], acc[mi][ni]);
    }
    if (it + 1 < K / 32 && wave < 2) __builtin_amdgcn_s_wait_tensorcnt(0);
    __syncthreads();
  }
#else
  __shared__ __align__(16) bf16_t sA[128 * LDA];
  __shared__ __align__(16) bf16_t sB[128 * LDA];
  for (int k0 = 0; k0 < K; k0 += 32) {
    __syncthreads();
#pragma unroll
    for (int i = 0; i < 2; ++i) {
      int c = tid + 256 * i;
      int r = c >> 2, cc = c & 3;
      *(float4*)(sA + r * LDA + cc * 8) =
          *(const float4*)(A + (size_t)(m0 + r) * K + k0 + cc * 8);
      *(float4*)(sB + r * LDA + cc * 8) =
          *(const float4*)(W + (size_t)(n0 + r) * K + k0 + cc * 8);
    }
    if (k0 + 32 < K) {
      __builtin_prefetch(A + (size_t)(m0 + (tid >> 1)) * K + k0 + 32, 0, 1);
      __builtin_prefetch(W + (size_t)(n0 + (tid >> 1)) * K + k0 + 32, 0, 1);
    }
    __syncthreads();
    v16bf bfrag[2];
#pragma unroll
    for (int ni = 0; ni < 2; ++ni)
      bfrag[ni] = load_frag_bt(sB + (wn * 32 + ni * 16) * LDA, LDA, lane);
#pragma unroll
    for (int mi = 0; mi < 4; ++mi) {
      v16bf afrag = load_frag_a(sA + (wm * 64 + mi * 16) * LDA, LDA, lane);
#pragma unroll
      for (int ni = 0; ni < 2; ++ni)
        acc[mi][ni] = wmma_bf16(afrag, bfrag[ni], acc[mi][ni]);
    }
  }
#endif

#pragma unroll
  for (int mi = 0; mi < 4; ++mi)
#pragma unroll
    for (int ni = 0; ni < 2; ++ni) {
      int n = n0 + wn * 32 + ni * 16 + (lane & 15);
      float bb = bias[n];
#pragma unroll
      for (int j = 0; j < 8; ++j) {
        int m = m0 + wm * 64 + mi * 16 + j + ((lane >> 4) << 3);
        Y[(size_t)m * NCOL + n] = (OutT)(acc[mi][ni][j] + bb);
      }
    }
}

// ---------------------------------------------------------------------------
// Kernel 4: flash attention. Block = (128 q-rows, head, batch), 8 waves,
// each wave owns 16 q-rows; K/V 64x64 tiles DMA'd to LDS via TDM.
// ---------------------------------------------------------------------------
__global__ __launch_bounds__(256) void flash_attn_kernel(
    const bf16_t* __restrict__ Q, const bf16_t* __restrict__ K,
    const bf16_t* __restrict__ V, bf16_t* __restrict__ O) {
  constexpr int D = 1024, HS = 64, Lctx = 2048, LDK = 72;  // 144B padded
  __shared__ __align__(16) bf16_t sK[64 * LDK];
  __shared__ __align__(16) bf16_t sV[64 * LDK];
  __shared__ __align__(16) bf16_t sP[8][16 * 72];

  int tid = threadIdx.x, lane = tid & 31, wave = tid >> 5;
  size_t base = (size_t)blockIdx.z * Lctx * D + (size_t)blockIdx.y * HS;
  int qrow0 = blockIdx.x * 128 + wave * 16;

  const bf16_t* qptr = Q + base + (size_t)qrow0 * D;
  v16bf qa[2];
  qa[0] = load_frag_a(qptr, D, lane);
  qa[1] = load_frag_a(qptr + 32, D, lane);

  v8f accO[4];
#pragma unroll
  for (int df = 0; df < 4; ++df) accO[df] = (v8f){};
  float mrow[8], lrow[8];
#pragma unroll
  for (int j = 0; j < 8; ++j) { mrow[j] = -1e30f; lrow[j] = 0.f; }
  const float scale = 0.125f * 1.44269504f;  // hs^-0.5 * log2(e)

  for (int kt = 0; kt < Lctx / 64; ++kt) {
    __syncthreads();
#if ATTN_USE_TDM
    // Tile rows: 64 elems = 128B -> pad_interval=4; pad 16B -> stride 72 = LDK.
    if (wave == 0)
      tdm_load_tile_2d(K + base + (size_t)(kt * 64) * D, lds_offset_of(sK),
                       D, 64, 64, 4, 3);
    else if (wave == 1)
      tdm_load_tile_2d(V + base + (size_t)(kt * 64) * D, lds_offset_of(sV),
                       D, 64, 64, 4, 3);
    if (wave < 2) __builtin_amdgcn_s_wait_tensorcnt(0);
#else
#pragma unroll
    for (int i = 0; i < 2; ++i) {
      int c = tid + 256 * i;
      int r = c >> 3, cc = c & 7;
      *(float4*)(sK + r * LDK + cc * 8) =
          *(const float4*)(K + base + (size_t)(kt * 64 + r) * D + cc * 8);
      *(float4*)(sV + r * LDK + cc * 8) =
          *(const float4*)(V + base + (size_t)(kt * 64 + r) * D + cc * 8);
    }
#endif
    __syncthreads();

    // S = Q * K^T  (16 x 64 per wave)
    v8f s[4];
#pragma unroll
    for (int nf = 0; nf < 4; ++nf) {
      s[nf] = (v8f){};
#pragma unroll
      for (int ks = 0; ks < 2; ++ks) {
        v16bf bf = load_frag_bt(sK + (nf * 16) * LDK + ks * 32, LDK, lane);
        s[nf] = wmma_bf16(qa[ks], bf, s[nf]);
      }
    }

    // Online softmax (log2 domain); row stats reduced across each half-wave.
#pragma unroll
    for (int j = 0; j < 8; ++j) {
      float mx = fmaxf(fmaxf(s[0][j], s[1][j]), fmaxf(s[2][j], s[3][j]));
#pragma unroll
      for (int off = 8; off >= 1; off >>= 1)
        mx = fmaxf(mx, __shfl_xor(mx, off, 32));
      float mnew  = fmaxf(mrow[j], mx * scale);
      float alpha = exp2f(mrow[j] - mnew);
      float rsum  = 0.f;
#pragma unroll
      for (int nf = 0; nf < 4; ++nf) {
        float p = exp2f(s[nf][j] * scale - mnew);
        s[nf][j] = p;
        rsum += p;
      }
#pragma unroll
      for (int off = 8; off >= 1; off >>= 1) rsum += __shfl_xor(rsum, off, 32);
      lrow[j] = lrow[j] * alpha + rsum;
      mrow[j] = mnew;
#pragma unroll
      for (int df = 0; df < 4; ++df) accO[df][j] *= alpha;
    }

    // D-layout -> row-major P tile (wave-private LDS), then A-frag reload.
    bf16_t* pw = &sP[wave][0];
#pragma unroll
    for (int nf = 0; nf < 4; ++nf) {
      int n = nf * 16 + (lane & 15);
#pragma unroll
      for (int j = 0; j < 8; ++j) {
        int m = j + ((lane >> 4) << 3);
        pw[m * 72 + n] = (bf16_t)s[nf][j];
      }
    }
    v16bf pa[2];
    pa[0] = load_frag_a(pw, 72, lane);
    pa[1] = load_frag_a(pw + 32, 72, lane);

    // accO += P * V
#pragma unroll
    for (int df = 0; df < 4; ++df) {
#pragma unroll
      for (int ks = 0; ks < 2; ++ks) {
        v16bf bv = load_frag_b(sV + ks * 32 * LDK + df * 16, LDK, lane);
        accO[df] = wmma_bf16(pa[ks], bv, accO[df]);
      }
    }
  }

  bf16_t* op = O + base + (size_t)qrow0 * D;
#pragma unroll
  for (int df = 0; df < 4; ++df) {
    int n = df * 16 + (lane & 15);
#pragma unroll
    for (int j = 0; j < 8; ++j) {
      int m = j + ((lane >> 4) << 3);
      op[(size_t)m * D + n] = (bf16_t)(accO[df][j] / lrow[j]);
    }
  }
}

// ---------------------------------------------------------------------------
// Host launcher. Inputs: x, ln_g, ln_b, Wq, bq, Wk, bk, Wv, bv, Wo, bo.
// Workspace (48 MiB): h | Wq..Wo(bf16) | Q | K | V | O.
// ---------------------------------------------------------------------------
extern "C" void kernel_launch(void* const* d_in, const int* in_sizes, int n_in,
                              void* d_out, int out_size, void* d_ws,
                              size_t ws_size, hipStream_t stream) {
  (void)in_sizes; (void)n_in; (void)out_size; (void)ws_size;
  const float* x   = (const float*)d_in[0];
  const float* lng = (const float*)d_in[1];
  const float* lnb = (const float*)d_in[2];
  const float* Wq  = (const float*)d_in[3];
  const float* bq  = (const float*)d_in[4];
  const float* Wk  = (const float*)d_in[5];
  const float* bk  = (const float*)d_in[6];
  const float* Wv  = (const float*)d_in[7];
  const float* bv  = (const float*)d_in[8];
  const float* Wo  = (const float*)d_in[9];
  const float* bo  = (const float*)d_in[10];
  float* out = (float*)d_out;

  char* ws = (char*)d_ws;
  bf16_t* hbf = (bf16_t*)(ws);
  bf16_t* wqb = (bf16_t*)(ws + (size_t)(8u  << 20));
  bf16_t* wkb = (bf16_t*)(ws + (size_t)(10u << 20));
  bf16_t* wvb = (bf16_t*)(ws + (size_t)(12u << 20));
  bf16_t* wob = (bf16_t*)(ws + (size_t)(14u << 20));
  bf16_t* Qb  = (bf16_t*)(ws + (size_t)(16u << 20));
  bf16_t* Kb  = (bf16_t*)(ws + (size_t)(24u << 20));
  bf16_t* Vb  = (bf16_t*)(ws + (size_t)(32u << 20));
  bf16_t* Ob  = (bf16_t*)(ws + (size_t)(40u << 20));

  ln_bf16_kernel<<<4096, 256, 0, stream>>>(x, lng, lnb, hbf);
  f32_to_bf16_kernel<<<1024, 256, 0, stream>>>(Wq, wqb, 1024 * 1024);
  f32_to_bf16_kernel<<<1024, 256, 0, stream>>>(Wk, wkb, 1024 * 1024);
  f32_to_bf16_kernel<<<1024, 256, 0, stream>>>(Wv, wvb, 1024 * 1024);
  f32_to_bf16_kernel<<<1024, 256, 0, stream>>>(Wo, wob, 1024 * 1024);

  dim3 gg(32, 8);  // 4096/128 x 1024/128
  gemm_bf16_kernel<bf16_t><<<gg, 256, 0, stream>>>(hbf, wqb, bq, Qb);
  gemm_bf16_kernel<bf16_t><<<gg, 256, 0, stream>>>(hbf, wkb, bk, Kb);
  gemm_bf16_kernel<bf16_t><<<gg, 256, 0, stream>>>(hbf, wvb, bv, Vb);

  dim3 ga(16, 16, 2);  // q-tiles x heads x batch
  flash_attn_kernel<<<ga, 256, 0, stream>>>(Qb, Kb, Vb, Ob);

  gemm_bf16_kernel<float><<<gg, 256, 0, stream>>>(Ob, wob, bo, out);
}
